// SSM_Block_1228360646850
// MI455X (gfx1250) — matrix operand order repaired and checked by reading, not compiled, not run on hardware
//
#include <hip/hip_runtime.h>
#include <hip/hip_bf16.h>

// ---------------------------------------------------------------------------
// SSM block for MI455X (gfx1250, wave32).
//
// The diagonal SSM with input-independent (A,B,C,D) is a per-channel causal
// convolution with a 64-tap kernel  k_c(tau) = sum_n C*B*sigmoid(A_log)^tau.
// So  Y_c (64 x 2048) = T_c (64x64 lower-triangular Toeplitz) * X_c (64x2048)
// run on v_wmma_f32_16x16x32_f16 (f16 in, fp32 accumulate).  D is folded into
// the Toeplitz diagonal.  LayerNorms and activations stay fp32; the GELU uses
// a branch-free erf (A&S 7.1.26, |err|<=1.5e-7) with hardware v_rcp_f32.
// ---------------------------------------------------------------------------

typedef __attribute__((ext_vector_type(16))) _Float16 v16h;
typedef __attribute__((ext_vector_type(8)))  _Float16 v8h;
typedef __attribute__((ext_vector_type(8)))  float    v8f;

#define SEQ     64
#define CH      256
#define NSTATE  64
#define NP      256      // spatial positions per frame (16*16)
#define BTOT    2048     // 8 clips * 256 spatial = GEMM N
#define FRAMES  512
#define EPS     1e-5f

// Single v_rcp_f32 (~1 ulp) instead of the IEEE div_scale/div_fixup sequence.
__device__ __forceinline__ float rcp_fast(float x)
{
    return __builtin_amdgcn_rcpf(x);
}

// Branch-free erf, Abramowitz & Stegun 7.1.26 (|err| <= 1.5e-7).
__device__ __forceinline__ float erf_fast(float x)
{
    float ax = fabsf(x);
    float t  = rcp_fast(1.f + 0.3275911f * ax);
    float p  = t * (0.254829592f +
               t * (-0.284496736f +
               t * (1.421413741f +
               t * (-1.453152027f +
               t *  1.061405429f))));
    float r  = 1.f - p * __expf(-ax * ax);
    return copysignf(r, x);
}

// ---------------------------------------------------------------------------
// Kernel 0: build per-channel conv kernel k_c(tau) and pack the 64x64
// Toeplitz matrix T_c (with D on the diagonal) in WMMA A-operand layout.
// A layout (ISA 7.12.2): lane L holds row M = L&15; value i (0..15) holds
// K = ((i<8)?0:16) + ((L<16)?0:8) + (i&7).
// Tiles stored as [mt(4)][kb(2)] of 512 f16 each -> 4096 f16 per channel.
// ---------------------------------------------------------------------------
__global__ void build_T(const float* __restrict__ A_log,
                        const float* __restrict__ Bm,
                        const float* __restrict__ Cm,
                        const float* __restrict__ Dv,
                        _Float16* __restrict__ Tpack)
{
    int c   = blockIdx.x;     // 256
    int tau = threadIdx.x;    // 64
    __shared__ float kk[SEQ];

    float acc = 0.f;
    for (int n = 0; n < NSTATE; ++n) {
        float al = A_log[c * NSTATE + n];
        float la = -log1pf(__expf(-al));          // log(sigmoid(al)), stable
        float w  = Cm[c * NSTATE + n] * Bm[c * NSTATE + n];
        acc += w * __expf((float)tau * la);       // C*B*Abar^tau
    }
    kk[tau] = acc + ((tau == 0) ? Dv[c] : 0.f);   // fold D into diagonal
    __syncthreads();

    _Float16* dst = Tpack + (size_t)c * 4096;
    for (int idx = threadIdx.x; idx < 4096; idx += 64) {
        int tile = idx >> 9;          // 0..7  -> (mt, kb)
        int pos  = idx & 511;
        int lane = pos >> 4;
        int i    = pos & 15;
        int mt = tile >> 1, kb = tile & 1;
        int m  = lane & 15;
        int k  = ((i < 8) ? 0 : 16) + ((lane < 16) ? 0 : 8) + (i & 7);
        int t  = mt * 16 + m;         // output time (row)
        int s  = kb * 32 + k;         // input time (col)
        float v = (s <= t) ? kk[t - s] : 0.f;
        dst[idx] = (_Float16)v;
    }
}

// ---------------------------------------------------------------------------
// Kernel 1: channel LayerNorm + pack X to f16 in [c][t][b] layout.
// One block per frame (a*64+t); thread p owns one spatial position, loops
// channels -> all global accesses are 128B-per-warp coalesced; no LDS needed.
// ix fits in the 192MB L2, so the second pass re-reads from cache.
// ---------------------------------------------------------------------------
__global__ void ln_in_pack(const float* __restrict__ ix,
                           const float* __restrict__ g_in,
                           const float* __restrict__ b_in,
                           _Float16* __restrict__ Xpack)
{
    int frame = blockIdx.x;       // 512
    int p     = threadIdx.x;      // 256
    const float* row0 = ix + (size_t)frame * CH * NP;

    float s = 0.f, sq = 0.f;
    for (int c = 0; c < CH; ++c) {
        float v = row0[(size_t)c * NP + p];
        s += v; sq += v * v;
    }
    float mean = s * (1.f / CH);
    float var  = sq * (1.f / CH) - mean * mean;
    float rstd = rsqrtf(var + EPS);

    int a = frame >> 6;           // clip index
    int t = frame & 63;           // time index
    for (int c = 0; c < CH; ++c) {
        float v  = row0[(size_t)c * NP + p];
        float xn = (v - mean) * rstd * g_in[c] + b_in[c];
        Xpack[((size_t)c * SEQ + t) * BTOT + a * NP + p] = (_Float16)xn;
    }
}

// ---------------------------------------------------------------------------
// Kernel 2: per-channel Toeplitz GEMM on WMMA + fp32 epilogue.
// Block = 128 threads (4 waves). Wave w computes M-tile rows 16w..16w+15 of
// Y_c for one N-tile of 16 batch columns. X tile (64t x 16b) is staged through
// LDS into B-operand layout (lanes 0-15 hold K=0..7,16..23, lanes 16-31 K=8..15,24..31,
// N = lane&15), so each lane reads one contiguous v16h per k-block.
// Epilogue: branch-free GELU(erf) then u*sigmoid(u); writes Y in the
// post-transform layout (f = a*64+t, c, p) straight into d_out (staging).
// ---------------------------------------------------------------------------
__global__ void ssm_wmma(const _Float16* __restrict__ Tpack,
                         const _Float16* __restrict__ Xpack,
                         float* __restrict__ Y)
{
    int ntile = blockIdx.x;       // 0..127 (N-tiles of 16 over BTOT)
    int c     = blockIdx.y;       // 0..255
    int tid   = threadIdx.x;      // 0..127
    int wave  = tid >> 5;
    int lane  = tid & 31;
    int b0    = ntile * 16;
    int a     = b0 >> 8;          // clip (tile never crosses a 256 boundary)
    int pbase = b0 & 255;

    __shared__ __align__(32) _Float16 xsh[2048];   // 2 kb * 1024 f16 (32 lanes * 16 f16 used per kb)

    // ---- stage X tile: coalesced 16B global loads, scattered LDS packing
    {
        int row  = tid >> 1;      // t: 0..63
        int half = tid & 1;       // which 8 of the 16 N columns
        v8h v = *(const v8h*)(Xpack + ((size_t)c * SEQ + row) * BTOT + b0 + half * 8);
        int kb = row >> 5;
        int r  = row & 31;
        int base = kb * 1024 + (r & 8) * 32 + (r & 7) + ((r & 16) >> 1);
#pragma unroll
        for (int j = 0; j < 8; ++j) {
            int ncol = half * 8 + j;
            xsh[base + ncol * 16] = v[j];
        }
    }
    __syncthreads();

    // ---- operands: A from packed global (L2-resident, 2MB total), B from LDS
    const _Float16* tA = Tpack + (size_t)c * 4096 + wave * 1024;
    v16h a0  = *(const v16h*)(tA + 0 * 512 + lane * 16);
    v16h a1  = *(const v16h*)(tA + 1 * 512 + lane * 16);
    v16h bx0 = *(const v16h*)(xsh + 0 * 1024 + lane * 16);
    v16h bx1 = *(const v16h*)(xsh + 1 * 1024 + lane * 16);

    v8f acc = {};
    acc = __builtin_amdgcn_wmma_f32_16x16x32_f16(false, a0, false, bx0,
                                                 (short)0, acc, false, false);
    acc = __builtin_amdgcn_wmma_f32_16x16x32_f16(false, a1, false, bx1,
                                                 (short)0, acc, false, false);

    // ---- fp32 epilogue (branch-free): GELU(erf), u*sigmoid(u); post-layout
    int n  = lane & 15;           // N column within tile
    int mh = (lane >> 4) * 8;     // accumulator row offset (C/D layout)
#pragma unroll
    for (int r = 0; r < 8; ++r) {
        int t = wave * 16 + r + mh;                             // 0..63
        float y = acc[r];
        float u = 0.5f * y * (1.f + erf_fast(y * 0.70710678118f)); // exact-form GELU
        float z = u * rcp_fast(1.f + __expf(-u));               // u*sigmoid(u)
        int f = a * SEQ + t;
        Y[((size_t)f * CH + c) * NP + pbase + n] = z;
    }
}

// ---------------------------------------------------------------------------
// Kernel 3: final spatial LayerNorms. One block per (f,c) row of 256.
// Reads Y row (staged in d_out) and ix row (L2-resident), writes result
// in-place -- safe: each block only touches its own row.
// Reductions: wave32 shfl-xor butterflies for all 4 statistics, one barrier.
// ---------------------------------------------------------------------------
__global__ void ln_out(const float* __restrict__ ix,
                       const float* __restrict__ g1, const float* __restrict__ b1,
                       const float* __restrict__ g2, const float* __restrict__ b2,
                       float* __restrict__ out)
{
    int row  = blockIdx.x;        // f*CH + c, 131072 rows
    int p    = threadIdx.x;       // 256
    int lane = p & 31;
    int wid  = p >> 5;            // 8 waves
    __shared__ float red[4][8];

    float z = out[(size_t)row * NP + p];
    float v = ix [(size_t)row * NP + p];

    float zs = z, zq = z * z, vs = v, vq = v * v;
#pragma unroll
    for (int m = 16; m > 0; m >>= 1) {
        zs += __shfl_xor(zs, m, 32);
        zq += __shfl_xor(zq, m, 32);
        vs += __shfl_xor(vs, m, 32);
        vq += __shfl_xor(vq, m, 32);
    }
    if (lane == 0) {
        red[0][wid] = zs; red[1][wid] = zq;
        red[2][wid] = vs; red[3][wid] = vq;
    }
    __syncthreads();
    zs = zq = vs = vq = 0.f;
#pragma unroll
    for (int w = 0; w < 8; ++w) {
        zs += red[0][w]; zq += red[1][w];
        vs += red[2][w]; vq += red[3][w];
    }

    float mz = zs * (1.f / NP);
    float rz = rsqrtf(zq * (1.f / NP) - mz * mz + EPS);
    float mv = vs * (1.f / NP);
    float rv = rsqrtf(vq * (1.f / NP) - mv * mv + EPS);

    float xf  = (z - mz) * rz * g1[p] + b1[p];
    float ixf = (v - mv) * rv * g2[p] + b2[p];
    out[(size_t)row * NP + p] = xf + ixf;
}

// ---------------------------------------------------------------------------
extern "C" void kernel_launch(void* const* d_in, const int* in_sizes, int n_in,
                              void* d_out, int out_size, void* d_ws, size_t ws_size,
                              hipStream_t stream)
{
    const float* ix    = (const float*)d_in[0];
    const float* A_log = (const float*)d_in[1];
    const float* B     = (const float*)d_in[2];
    const float* C     = (const float*)d_in[3];
    const float* D     = (const float*)d_in[4];
    const float* g_in  = (const float*)d_in[5];
    const float* b_in  = (const float*)d_in[6];
    const float* g1    = (const float*)d_in[7];
    const float* b1    = (const float*)d_in[8];
    const float* g2    = (const float*)d_in[9];
    const float* b2    = (const float*)d_in[10];
    float* out = (float*)d_out;

    // Workspace: Tpack (2 MB) + Xpack (64 MB); Y staged in d_out itself.
    _Float16* Tpack = (_Float16*)d_ws;
    _Float16* Xpack = (_Float16*)((char*)d_ws + (size_t)CH * 4096 * sizeof(_Float16));

    build_T   <<<dim3(CH),            dim3(64),  0, stream>>>(A_log, B, C, D, Tpack);
    ln_in_pack<<<dim3(FRAMES),        dim3(256), 0, stream>>>(ix, g_in, b_in, Xpack);
    ssm_wmma  <<<dim3(BTOT / 16, CH), dim3(128), 0, stream>>>(Tpack, Xpack, out);
    ln_out    <<<dim3(FRAMES * CH),   dim3(256), 0, stream>>>(ix, g1, b1, g2, b2, out);
}
